// ProbSparseAttention_73985106641345
// MI455X (gfx1250) — matrix operand order repaired and checked by reading, not compiled, not run on hardware
//
#include <hip/hip_runtime.h>
#include <hip/hip_bf16.h>
#include <math.h>

// ---------------- constants (match reference) ----------------
constexpr int BATCH   = 8;
constexpr int SEQ     = 4096;
constexpr int DM      = 512;
constexpr int NH      = 8;
constexpr int HD      = 64;
constexpr int MROWS   = BATCH * SEQ;   // 32768

typedef __bf16 v16bf __attribute__((ext_vector_type(16)));
typedef float  v8f   __attribute__((ext_vector_type(8)));

static __device__ inline v8f wmma_bf16(v16bf a, v16bf b, v8f c) {
  return __builtin_amdgcn_wmma_f32_16x16x32_bf16(false, a, false, b, (short)0, c, false, false);
}

union FragU { v16bf v; uint4 q[2]; };

// =====================================================================
// Kernel 1: C[M,512] = X[M,512] @ W[512,512]^T + bias, bf16 WMMA.
// mode 0: fp32 row-major [M,512]
// mode 1: bf16 [B,H,L,D]   (Q, K)
// mode 2: bf16 [B,H,D,L]   (V transposed)
// =====================================================================
__global__ __launch_bounds__(256)
void gemm_wmma_kernel(const float* __restrict__ X, const float* __restrict__ W,
                      const float* __restrict__ bias, void* __restrict__ out, int mode)
{
  constexpr int K  = DM;
  constexpr int LP = 48;                      // padded LDS row (96B, 16B-aligned)
  __shared__ __bf16 Xs[128][LP];
  __shared__ __bf16 Ws[128][LP];

  const int tid  = threadIdx.x;
  const int wave = tid >> 5;
  const int lane = tid & 31;
  const int bm   = blockIdx.y * 128;
  const int bn   = blockIdx.x * 128;
  const int mw   = (wave & 3) * 32;           // wave M offset (4 waves over M)
  const int nw   = (wave >> 2) * 64;          // wave N offset (2 waves over N)

  v8f acc[2][4];
#pragma unroll
  for (int i = 0; i < 2; i++)
#pragma unroll
    for (int j = 0; j < 4; j++) acc[i][j] = (v8f){0.f,0.f,0.f,0.f,0.f,0.f,0.f,0.f};

  const int lr  = tid >> 1;                   // 0..127 load row
  const int lc0 = (tid & 1) * 16;             // 0 or 16

  for (int kb = 0; kb < K; kb += 32) {
    // ---- stage X and W tiles into LDS as bf16 ----
    const float* xp = X + (size_t)(bm + lr) * K + kb + lc0;
    const float* wp = W + (size_t)(bn + lr) * K + kb + lc0;
    __builtin_prefetch(xp + 32, 0, 1);        // global_prefetch_b8: next K-block
    __builtin_prefetch(wp + 32, 0, 1);
#pragma unroll
    for (int i = 0; i < 4; i++) {
      float4 vx = ((const float4*)xp)[i];
      float4 vw = ((const float4*)wp)[i];
      union { __bf16 h[4]; unsigned long long u; } px, pw;
      px.h[0] = (__bf16)vx.x; px.h[1] = (__bf16)vx.y; px.h[2] = (__bf16)vx.z; px.h[3] = (__bf16)vx.w;
      pw.h[0] = (__bf16)vw.x; pw.h[1] = (__bf16)vw.y; pw.h[2] = (__bf16)vw.z; pw.h[3] = (__bf16)vw.w;
      *(unsigned long long*)&Xs[lr][lc0 + i * 4] = px.u;
      *(unsigned long long*)&Ws[lr][lc0 + i * 4] = pw.u;
    }
    __syncthreads();

    // ---- A fragments (16x32 bf16, ISA lane layout) ----
    FragU afr[2];
#pragma unroll
    for (int mt = 0; mt < 2; mt++) {
      const __bf16* ap = &Xs[mw + mt * 16 + (lane & 15)][0];
      const int c0 = (lane >> 4) * 8;
      afr[mt].q[0] = *(const uint4*)(ap + c0);
      afr[mt].q[1] = *(const uint4*)(ap + c0 + 16);
    }
    // ---- B fragments (32x16 bf16: column = W row) + WMMA ----
#pragma unroll
    for (int nt = 0; nt < 4; nt++) {
      FragU bfr;
      const __bf16* bp = &Ws[nw + nt * 16 + (lane & 15)][0];
      const int kk = (lane >> 4) * 8;
      bfr.q[0] = *(const uint4*)(bp + kk);
      bfr.q[1] = *(const uint4*)(bp + kk + 16);
#pragma unroll
      for (int mt = 0; mt < 2; mt++)
        acc[mt][nt] = wmma_bf16(afr[mt].v, bfr.v, acc[mt][nt]);
    }
    __syncthreads();
  }

  // ---- epilogue: bias + scatter per output mode ----
  const int rowloc = (lane >> 4) * 8;
#pragma unroll
  for (int nt = 0; nt < 4; nt++) {
    const int n  = bn + nw + nt * 16 + (lane & 15);
    const float bv = bias[n];
#pragma unroll
    for (int mt = 0; mt < 2; mt++) {
#pragma unroll
      for (int j = 0; j < 8; j++) {
        const int m = bm + mw + mt * 16 + rowloc + j;
        const float val = acc[mt][nt][j] + bv;
        if (mode == 0) {
          ((float*)out)[(size_t)m * DM + n] = val;
        } else {
          const int b = m >> 12, l = m & (SEQ - 1);
          const int h = n >> 6,  d = n & (HD - 1);
          if (mode == 1)
            ((__bf16*)out)[((((size_t)b * NH + h) * SEQ + l) << 6) + d] = (__bf16)val;
          else
            ((__bf16*)out)[(((size_t)b * NH + h) * HD + d) * SEQ + l] = (__bf16)val;
        }
      }
    }
  }
}

// =====================================================================
// Kernel 2: sampled scores  M[b,h,l] = scale*(max_s q.k_s - mean_s q.k_s)
// =====================================================================
__global__ __launch_bounds__(256)
void sample_scores_kernel(const __bf16* __restrict__ Q, const __bf16* __restrict__ Kb,
                          const long long* __restrict__ sidx, float* __restrict__ Mout,
                          int u, float scale)
{
  __shared__ float Ks[64][HD];
  const int bh = blockIdx.y;
  const int tid = threadIdx.x;
  for (int e = tid; e < u * HD; e += 256) {
    const int s = e >> 6, d = e & (HD - 1);
    Ks[s][d] = (float)Kb[((size_t)bh * SEQ + (int)sidx[s]) * HD + d];
  }
  __syncthreads();

  const int l = blockIdx.x * 256 + tid;
  const __bf16* qp = Q + ((size_t)bh * SEQ + l) * HD;
  float q[HD];
#pragma unroll
  for (int d = 0; d < HD; d++) q[d] = (float)qp[d];

  float mx = -3.4e38f, sum = 0.f;
  for (int s = 0; s < u; s++) {
    float acc = 0.f;
#pragma unroll
    for (int d = 0; d < HD; d++) acc += q[d] * Ks[s][d];
    mx = fmaxf(mx, acc);
    sum += acc;
  }
  Mout[(size_t)bh * SEQ + l] = (mx - sum / (float)u) * scale;
}

// =====================================================================
// Kernel 3: iterative top-u argmax per (b,h)  (u <= 64)
// =====================================================================
__global__ __launch_bounds__(256)
void topk_kernel(const float* __restrict__ Mbuf, int* __restrict__ topi, int u)
{
  __shared__ float vals[SEQ];
  __shared__ float rv[256];
  __shared__ int   ri[256];
  const int bh = blockIdx.x, tid = threadIdx.x;
  for (int i = tid; i < SEQ; i += 256) vals[i] = Mbuf[(size_t)bh * SEQ + i];
  __syncthreads();

  for (int it = 0; it < u; it++) {
    float bv = -3.4e38f; int bi = 0;
    for (int i = tid; i < SEQ; i += 256) {
      const float v = vals[i];
      if (v > bv) { bv = v; bi = i; }
    }
    rv[tid] = bv; ri[tid] = bi;
    __syncthreads();
    for (int s = 128; s > 0; s >>= 1) {
      if (tid < s) {
        if (rv[tid + s] > rv[tid] || (rv[tid + s] == rv[tid] && ri[tid + s] < ri[tid])) {
          rv[tid] = rv[tid + s]; ri[tid] = ri[tid + s];
        }
      }
      __syncthreads();
    }
    if (tid == 0) { topi[bh * 64 + it] = ri[0]; vals[ri[0]] = -3.4e38f; }
    __syncthreads();
  }
}

// =====================================================================
// Kernel 4: V mean over L (from transposed V) ; Kernel 5: broadcast fill
// =====================================================================
__global__ __launch_bounds__(256)
void vmean_kernel(const __bf16* __restrict__ Vt, float* __restrict__ mean)
{
  const int t = blockIdx.x * 256 + threadIdx.x;     // bh*64 + d  (== b*512 + h*64 + d)
  const __bf16* p = Vt + (size_t)t * SEQ;
  float s = 0.f;
  for (int l = 0; l < SEQ; l++) s += (float)p[l];
  mean[t] = s * (1.f / (float)SEQ);
}

__global__ __launch_bounds__(256)
void fill_mean_kernel(const float* __restrict__ mean, float* __restrict__ out_attn)
{
  const size_t total = (size_t)BATCH * SEQ * DM;
  for (size_t i = (size_t)blockIdx.x * blockDim.x + threadIdx.x; i < total;
       i += (size_t)gridDim.x * blockDim.x) {
    const int n = (int)(i & (DM - 1));
    const int b = (int)(i >> 21);                   // / (4096*512)
    out_attn[i] = mean[b * DM + n];
  }
}

// =====================================================================
// Kernel 6: flash-style WMMA attention for the u reduced queries.
// One block per (b,h); wave w owns query tile w (rows w*16..w*16+15).
// =====================================================================
__global__ __launch_bounds__(128)
void attn_wmma_kernel(const __bf16* __restrict__ Qb, const __bf16* __restrict__ Kb,
                      const __bf16* __restrict__ Vt, const int* __restrict__ topi,
                      float* __restrict__ out_attn, int u, float scale)
{
  __shared__ __bf16 Qs[64][HD];
  __shared__ __bf16 Pbuf[4][16][32];
  __shared__ int    sidx[64];

  const int bh = blockIdx.x;
  const int b  = bh / NH, h = bh % NH;
  const int tid  = threadIdx.x;
  const int wave = tid >> 5;
  const int lane = tid & 31;
  const int nrows = (int)blockDim.x >> 1;           // UT*16

  for (int i = tid; i < u; i += blockDim.x) sidx[i] = topi[bh * 64 + i];
  __syncthreads();
  for (int e = tid; e < nrows * HD; e += blockDim.x) {
    const int r = e >> 6, d = e & (HD - 1);
    float qv = 0.f;
    if (r < u) qv = (float)Qb[((size_t)bh * SEQ + sidx[r]) * HD + d] * scale;
    Qs[r][d] = (__bf16)qv;
  }
  __syncthreads();

  // Q_red A-fragments for K-dim blocks [0,32) and [32,64)
  FragU aq0, aq1;
  {
    const __bf16* qp = &Qs[wave * 16 + (lane & 15)][0];
    const int c0 = (lane >> 4) * 8;
    aq0.q[0] = *(const uint4*)(qp + c0);
    aq0.q[1] = *(const uint4*)(qp + c0 + 16);
    aq1.q[0] = *(const uint4*)(qp + 32 + c0);
    aq1.q[1] = *(const uint4*)(qp + 32 + c0 + 16);
  }

  v8f Ofr[4];
#pragma unroll
  for (int t = 0; t < 4; t++) Ofr[t] = (v8f){0.f,0.f,0.f,0.f,0.f,0.f,0.f,0.f};
  float mrow[8], lrow[8];
#pragma unroll
  for (int j = 0; j < 8; j++) { mrow[j] = -3.0e38f; lrow[j] = 0.f; }

  const __bf16* Kbase = Kb + (size_t)bh * SEQ * HD;
  const __bf16* Vbase = Vt + (size_t)bh * HD * SEQ;

  for (int lc = 0; lc < SEQ; lc += 32) {
#pragma unroll
    for (int half = 0; half < 2; half++) {
      const int l0 = lc + half * 16;
      const __bf16* kp = Kbase + (size_t)(l0 + (lane & 15)) * HD;
      const int kk = (lane >> 4) * 8;
      FragU b0, b1;
      b0.q[0] = *(const uint4*)(kp + kk);
      b0.q[1] = *(const uint4*)(kp + kk + 16);
      b1.q[0] = *(const uint4*)(kp + 32 + kk);
      b1.q[1] = *(const uint4*)(kp + 32 + kk + 16);

      v8f s = (v8f){0.f,0.f,0.f,0.f,0.f,0.f,0.f,0.f};
      s = wmma_bf16(aq0.v, b0.v, s);
      s = wmma_bf16(aq1.v, b1.v, s);

      // online softmax: rows live per (vgpr j, lane-half); columns stripe 16 lanes
#pragma unroll
      for (int j = 0; j < 8; j++) {
        const float v = s[j];
        float rm = v;
        rm = fmaxf(rm, __shfl_xor(rm, 1, 32));
        rm = fmaxf(rm, __shfl_xor(rm, 2, 32));
        rm = fmaxf(rm, __shfl_xor(rm, 4, 32));
        rm = fmaxf(rm, __shfl_xor(rm, 8, 32));
        const float nm    = fmaxf(mrow[j], rm);
        const float alpha = __expf(mrow[j] - nm);
        const float p     = __expf(v - nm);
        float rs = p;
        rs += __shfl_xor(rs, 1, 32);
        rs += __shfl_xor(rs, 2, 32);
        rs += __shfl_xor(rs, 4, 32);
        rs += __shfl_xor(rs, 8, 32);
        lrow[j] = lrow[j] * alpha + rs;
        mrow[j] = nm;
#pragma unroll
        for (int t = 0; t < 4; t++) Ofr[t][j] *= alpha;
        Pbuf[wave][j + 8 * (lane >> 4)][half * 16 + (lane & 15)] = (__bf16)p;
      }
    }
    __syncthreads();

    // P (16x32 bf16) as A-fragment; V^T rows give contiguous B-fragments
    FragU pf;
    {
      const __bf16* pp = &Pbuf[wave][lane & 15][0];
      const int c0 = (lane >> 4) * 8;
      pf.q[0] = *(const uint4*)(pp + c0);
      pf.q[1] = *(const uint4*)(pp + c0 + 16);
    }
#pragma unroll
    for (int t = 0; t < 4; t++) {
      const __bf16* vp = Vbase + (size_t)(t * 16 + (lane & 15)) * SEQ + lc + (lane >> 4) * 8;
      FragU vf;
      vf.q[0] = *(const uint4*)vp;
      vf.q[1] = *(const uint4*)(vp + 16);
      Ofr[t] = wmma_bf16(pf.v, vf.v, Ofr[t]);
    }
    __syncthreads();
  }

  // scatter out_red rows into out_attn at top_idx positions
#pragma unroll
  for (int j = 0; j < 8; j++) {
    const int r = wave * 16 + j + 8 * (lane >> 4);
    if (r < u) {
      const int l = sidx[r];
      const float inv = 1.f / lrow[j];
#pragma unroll
      for (int t = 0; t < 4; t++) {
        const int d = t * 16 + (lane & 15);
        out_attn[((size_t)b * SEQ + l) * DM + h * HD + d] = Ofr[t][j] * inv;
      }
    }
  }
}

// =====================================================================
// host launch
// =====================================================================
extern "C" void kernel_launch(void* const* d_in, const int* in_sizes, int n_in,
                              void* d_out, int out_size, void* d_ws, size_t ws_size,
                              hipStream_t stream) {
  const float* x  = (const float*)d_in[0];
  const float* Wq = (const float*)d_in[1];
  const float* bq = (const float*)d_in[2];
  const float* Wk = (const float*)d_in[3];
  const float* bk = (const float*)d_in[4];
  const float* Wv = (const float*)d_in[5];
  const float* bv = (const float*)d_in[6];
  const float* Wo = (const float*)d_in[7];
  const float* bo = (const float*)d_in[8];
  const long long* sampleIdx = (const long long*)d_in[9];

  int u = in_sizes[9];
  if (u > 64) u = 64;
  const int UT = (u + 15) / 16;
  const float scale = 1.0f / sqrtf((float)HD);

  char* ws = (char*)d_ws;
  __bf16* Qb  = (__bf16*)ws; ws += (size_t)BATCH * NH * SEQ * HD * 2;   // 32 MB
  __bf16* Kb  = (__bf16*)ws; ws += (size_t)BATCH * NH * SEQ * HD * 2;   // 32 MB
  __bf16* Vtb = (__bf16*)ws; ws += (size_t)BATCH * NH * SEQ * HD * 2;   // 32 MB
  float* out_attn = (float*)ws; ws += (size_t)BATCH * SEQ * DM * 4;     // 64 MB
  float* Mbuf     = (float*)ws; ws += (size_t)BATCH * NH * SEQ * 4;     //  1 MB
  float* meanb    = (float*)ws; ws += (size_t)BATCH * NH * HD * 4;
  int*   topi     = (int*)ws;

  dim3 gGrid(DM / 128, MROWS / 128);

  // Q / K / V projections (bf16 outputs; V transposed for AV WMMA)
  gemm_wmma_kernel<<<gGrid, 256, 0, stream>>>(x, Wq, bq, (void*)Qb, 1);
  gemm_wmma_kernel<<<gGrid, 256, 0, stream>>>(x, Wk, bk, (void*)Kb, 1);
  gemm_wmma_kernel<<<gGrid, 256, 0, stream>>>(x, Wv, bv, (void*)Vtb, 2);

  // sampled scores -> sparsity measure M
  sample_scores_kernel<<<dim3(SEQ / 256, BATCH * NH), 256, 0, stream>>>(
      Qb, Kb, sampleIdx, Mbuf, u, scale);

  // top-u indices per (b,h)
  topk_kernel<<<BATCH * NH, 256, 0, stream>>>(Mbuf, topi, u);

  // V mean + broadcast baseline output
  vmean_kernel<<<(BATCH * NH * HD) / 256, 256, 0, stream>>>(Vtb, meanb);
  fill_mean_kernel<<<8192, 256, 0, stream>>>(meanb, out_attn);

  // reduced-query flash attention, scatter into out_attn
  attn_wmma_kernel<<<BATCH * NH, UT * 32, 0, stream>>>(
      Qb, Kb, Vtb, topi, out_attn, u, scale);

  // output projection -> d_out (fp32, [B*L, 512])
  gemm_wmma_kernel<<<gGrid, 256, 0, stream>>>(out_attn, Wo, bo, d_out, 0);
}
